// Non_local_Block_11673721111176
// MI455X (gfx1250) — matrix operand order repaired and checked by reading, not compiled, not run on hardware
//
#include <hip/hip_runtime.h>
#include <math.h>

// ---------------- WMMA types & helpers (CDNA5 gfx1250, wave32) ----------------
typedef __attribute__((ext_vector_type(16))) __bf16 v16bf;
typedef __attribute__((ext_vector_type(8)))  float  v8f;

union FragB {                 // 16 bf16 elements = 32 bytes = 8 VGPRs/lane
    v16bf v;
    uint4 q[2];
    unsigned short s[16];
};

union Pack8 { uint4 v; unsigned short s[8]; };

__device__ __forceinline__ unsigned short f2bf(float f) {
    unsigned int u = __builtin_bit_cast(unsigned int, f);
    unsigned int r = u + 0x7FFFu + ((u >> 16) & 1u);   // round-to-nearest-even
    return (unsigned short)(r >> 16);
}

__device__ __forceinline__ v8f wmma_bf16(const FragB& a, const FragB& b, v8f c) {
    // D = A(16x32 bf16) * B(32x16 bf16) + C(16x16 f32)
    return __builtin_amdgcn_wmma_f32_16x16x32_bf16(
        /*neg_a=*/false, a.v, /*neg_b=*/false, b.v,
        /*c_mod=*/(short)0, c, /*reuse_a=*/false, /*reuse_b=*/false);
}

// Async Global->LDS staging (gfx1250 ASYNCcnt path), guarded so absence of the
// builtin falls back to the proven direct-global version.
#if defined(__has_builtin)
#if __has_builtin(__builtin_amdgcn_global_load_async_to_lds_b128) && \
    __has_builtin(__builtin_amdgcn_s_wait_asynccnt)
#define USE_ASYNC_LDS 1
#endif
#endif

#ifdef USE_ASYNC_LDS
typedef int b128v __attribute__((vector_size(16)));   // matches builtin param type
__device__ __forceinline__ void async_cp16(const unsigned short* g, unsigned short* l) {
    __builtin_amdgcn_global_load_async_to_lds_b128(
        (__attribute__((address_space(1))) b128v*)g,
        (__attribute__((address_space(3))) b128v*)l, 0, 0);
}
#endif

// Problem constants
#define BB   4
#define CIN  256
#define CO   128
#define NN   4096    // 64*64

// ---------------- Kernel 1: transpose + convert x -> xt bf16 [B][N][CIN] ------
__global__ void k_xt(const float* __restrict__ x, unsigned short* __restrict__ xt) {
    __shared__ unsigned short t[32][33];
    const int b  = blockIdx.z;
    const int c0 = blockIdx.y * 32;
    const int n0 = blockIdx.x * 32;
    const int tx = threadIdx.x, ty = threadIdx.y;       // block (32,8)
#pragma unroll
    for (int r = 0; r < 4; ++r) {
        int cl = ty * 4 + r;
        t[tx][cl] = f2bf(x[((size_t)(b * CIN + c0 + cl)) * NN + n0 + tx]);
    }
    __syncthreads();
#pragma unroll
    for (int r = 0; r < 4; ++r) {
        int nl = ty * 4 + r;
        xt[((size_t)(b * NN + n0 + nl)) * CIN + c0 + tx] = t[nl][tx];
    }
}

// ---------------- Kernel 2: convert 4 weight matrices to bf16 -----------------
__global__ void k_wcvt(const float* __restrict__ a, const float* __restrict__ b,
                       const float* __restrict__ c, const float* __restrict__ d,
                       unsigned short* __restrict__ oa, unsigned short* __restrict__ ob,
                       unsigned short* __restrict__ oc, unsigned short* __restrict__ od) {
    int i = blockIdx.x * 256 + threadIdx.x;             // 128 blocks -> 32768 = 128*256
    if (i < CO * CIN) {
        oa[i] = f2bf(a[i]); ob[i] = f2bf(b[i]);
        oc[i] = f2bf(c[i]); od[i] = f2bf(d[i]);
    }
}

// ---------------- Kernel 3: QKV projections via WMMA --------------------------
// q,k stored [B][N][CO] bf16 ; v stored transposed [B][CO][N] bf16
__global__ void __launch_bounds__(128) k_proj(
        const unsigned short* __restrict__ xt,
        const unsigned short* __restrict__ wq, const unsigned short* __restrict__ wk,
        const unsigned short* __restrict__ wv,
        const float* __restrict__ bq, const float* __restrict__ bk,
        const float* __restrict__ bv,
        unsigned short* __restrict__ q, unsigned short* __restrict__ k,
        unsigned short* __restrict__ vt) {
    const int b   = blockIdx.y;
    const int n0  = blockIdx.x * 16;
    const int lane = threadIdx.x & 31;
    const int wid  = threadIdx.x >> 5;
    const int col  = lane & 15;
    const int hi   = lane >> 4;

    const unsigned short* ws[3] = {wq, wk, wv};
    const float* bs[3]          = {bq, bk, bv};

    v8f zero = {0.f,0.f,0.f,0.f,0.f,0.f,0.f,0.f};
    v8f acc[3][2];
#pragma unroll
    for (int p = 0; p < 3; ++p) { acc[p][0] = zero; acc[p][1] = zero; }

    const unsigned short* xrow = xt + ((size_t)(b * NN + n0 + col)) * CIN;
#pragma unroll
    for (int cc = 0; cc < 8; ++cc) {                    // K = 256 in 8 chunks of 32
        FragB bx;                                       // B-frag: col=n, K=channels contiguous
        const unsigned short* p = xrow + cc * 32 + hi * 16;
        bx.q[0] = *(const uint4*)p; bx.q[1] = *(const uint4*)(p + 8);
#pragma unroll
        for (int pi = 0; pi < 3; ++pi)
#pragma unroll
            for (int t = 0; t < 2; ++t) {
                const int o0 = wid * 32 + t * 16;       // output-channel tile base
                const unsigned short* ap = ws[pi] + (size_t)(o0 + col) * CIN + cc * 32 + hi * 16;
                FragB a;                                // A-frag: row=o, K contiguous 16h..16h+15
                a.q[0] = *(const uint4*)ap; a.q[1] = *(const uint4*)(ap + 8);
                acc[pi][t] = wmma_bf16(a, bx, acc[pi][t]);
            }
    }
#pragma unroll
    for (int pi = 0; pi < 3; ++pi)
#pragma unroll
        for (int t = 0; t < 2; ++t) {
            const int oe = wid * 32 + t * 16 + hi * 8;  // first D row for this lane-half
            const float4 b0 = *(const float4*)(bs[pi] + oe);
            const float4 b1 = *(const float4*)(bs[pi] + oe + 4);
            float bias[8] = {b0.x,b0.y,b0.z,b0.w,b1.x,b1.y,b1.z,b1.w};
            if (pi < 2) {
                Pack8 o8;
#pragma unroll
                for (int r = 0; r < 8; ++r) o8.s[r] = f2bf(acc[pi][t][r] + bias[r]);
                unsigned short* dst = (pi == 0 ? q : k) + ((size_t)(b * NN + n0 + col)) * CO + oe;
                *(uint4*)dst = o8.v;
            } else {
#pragma unroll
                for (int r = 0; r < 8; ++r)
                    vt[((size_t)(b * CO + oe + r)) * NN + n0 + col] = f2bf(acc[2][t][r] + bias[r]);
            }
        }
}

// ---------------- Kernel 4: fused flash attention -----------------------------
// One wave handles 16 query rows; the 4 waves of a block share the same batch
// and key schedule, so K/V blocks are staged once per block in LDS via the
// async Global->LDS DMA path (double buffered), when available.
__global__ void __launch_bounds__(128) k_attn(
        const unsigned short* __restrict__ q, const unsigned short* __restrict__ k,
        const unsigned short* __restrict__ vt, unsigned short* __restrict__ y) {
    __shared__ float Sb[4][16][32];
    __shared__ __align__(16) unsigned short Pb[4][16][32];
    __shared__ float scb[4][16];
    __shared__ float invb[4][16];
#ifdef USE_ASYNC_LDS
    __shared__ __align__(16) unsigned short Kl[2][32][128];   // [buf][key][ch]
    __shared__ __align__(16) unsigned short Vl[2][128][32];   // [buf][ch][key]
#endif

    const int wid  = threadIdx.x >> 5;
    const int lane = threadIdx.x & 31;
    const int col  = lane & 15;
    const int hi   = lane >> 4;
    const int gw   = blockIdx.x * 4 + wid;              // 1024 waves total
    const int b    = gw >> 8;
    const int q0   = (gw & 255) * 16;

    const unsigned short* kbase = k  + (size_t)b * NN * CO;
    const unsigned short* vbase = vt + (size_t)b * CO * NN;

    // resident Q fragments: A-layout, K = 128 channels in 4 chunks
    FragB Qf[4];
    const unsigned short* qrow = q + ((size_t)(b * NN) + q0 + col) * CO;
#pragma unroll
    for (int c = 0; c < 4; ++c) {
        const unsigned short* p = qrow + c * 32 + hi * 16;
        Qf[c].q[0] = *(const uint4*)p;
        Qf[c].q[1] = *(const uint4*)(p + 8);
    }

    v8f zero = {0.f,0.f,0.f,0.f,0.f,0.f,0.f,0.f};
    v8f acc[8];
#pragma unroll
    for (int t = 0; t < 8; ++t) acc[t] = zero;
    float m_run = -__builtin_inff();
    float s_run = 0.f;

#ifdef USE_ASYNC_LDS
    const int tid = threadIdx.x;
    // prologue: stage first K/V block into buffer 0 (8 async b128 per thread)
    {
#pragma unroll
        for (int r = 0; r < 4; ++r) {                   // K: 512 chunks of 16B
            int c = tid + 128 * r;
            async_cp16(kbase + (size_t)(c >> 4) * CO + (c & 15) * 8,
                       &Kl[0][c >> 4][(c & 15) * 8]);
        }
#pragma unroll
        for (int r = 0; r < 4; ++r) {                   // V: 512 chunks of 16B
            int c = tid + 128 * r;
            async_cp16(vbase + (size_t)(c >> 2) * NN + (c & 3) * 8,
                       &Vl[0][c >> 2][(c & 3) * 8]);
        }
    }
#endif

    for (int j0 = 0; j0 < NN; j0 += 32) {
#ifdef USE_ASYNC_LDS
        const int  cur = (j0 >> 5) & 1;
        const bool hn  = (j0 + 32) < NN;
        if (hn) {                                       // stage next block into 1-cur
            const int jn = j0 + 32;
#pragma unroll
            for (int r = 0; r < 4; ++r) {
                int c = tid + 128 * r;
                async_cp16(kbase + (size_t)(jn + (c >> 4)) * CO + (c & 15) * 8,
                           &Kl[1 - cur][c >> 4][(c & 15) * 8]);
            }
#pragma unroll
            for (int r = 0; r < 4; ++r) {
                int c = tid + 128 * r;
                async_cp16(vbase + (size_t)(c >> 2) * NN + jn + (c & 3) * 8,
                           &Vl[1 - cur][c >> 2][(c & 3) * 8]);
            }
            __builtin_amdgcn_s_wait_asynccnt(8);        // current buffer complete
        } else {
            __builtin_amdgcn_s_wait_asynccnt(0);
        }
        __syncthreads();
#endif
        // ---- S = Q K^T for two 16-key subtiles ----
#pragma unroll
        for (int s = 0; s < 2; ++s) {
            v8f S = zero;
#ifdef USE_ASYNC_LDS
            const unsigned short* krow = &Kl[cur][s * 16 + col][hi * 16];
#else
            const unsigned short* krow = kbase + (size_t)(j0 + s * 16 + col) * CO + hi * 16;
#endif
#pragma unroll
            for (int c = 0; c < 4; ++c) {               // B-frag: col=key, K=channels contiguous
                FragB kf;
                const unsigned short* p = krow + c * 32;
                kf.q[0] = *(const uint4*)p; kf.q[1] = *(const uint4*)(p + 8);
                S = wmma_bf16(Qf[c], kf, S);
            }
#pragma unroll
            for (int r = 0; r < 8; ++r) Sb[wid][r + 8 * hi][s * 16 + col] = S[r];
        }
#ifndef USE_ASYNC_LDS
        if (j0 + 32 < NN)                               // prefetch next K block (L2 resident)
            __builtin_prefetch(kbase + (size_t)(j0 + 32 + col) * CO, 0, 2);
#endif
        __syncthreads();

        // ---- online softmax: lanes 0..15 each own one query row ----
        if (lane < 16) {
            const int row = lane;
            float mx = m_run;
#pragma unroll
            for (int j = 0; j < 32; ++j) mx = fmaxf(mx, Sb[wid][row][j]);
            float scale = __expf(m_run - mx);
            float sum   = s_run * scale;
#pragma unroll
            for (int j = 0; j < 32; ++j) {
                float e = __expf(Sb[wid][row][j] - mx);
                sum += e;
                Pb[wid][row][j] = f2bf(e);
            }
            m_run = mx; s_run = sum;
            scb[wid][row] = scale;
        }
        __syncthreads();

        // ---- rescale accumulators, O += P V ----
        float sc[8];
#pragma unroll
        for (int r = 0; r < 8; ++r) sc[r] = scb[wid][r + 8 * hi];

        FragB Pf;                                       // A-layout read of 16x32 P tile
        {
            const unsigned short* pp = &Pb[wid][col][hi * 16];
            Pf.q[0] = *(const uint4*)pp;
            Pf.q[1] = *(const uint4*)(pp + 8);
        }
#pragma unroll
        for (int t = 0; t < 8; ++t) {
#pragma unroll
            for (int r = 0; r < 8; ++r) acc[t][r] *= sc[r];
            FragB vf;                                   // B-frag: col=out-channel, K=keys contiguous
#ifdef USE_ASYNC_LDS
            const unsigned short* vp = &Vl[cur][t * 16 + col][hi * 16];
#else
            const unsigned short* vp = vbase + (size_t)(t * 16 + col) * NN + j0 + hi * 16;
#endif
            vf.q[0] = *(const uint4*)vp; vf.q[1] = *(const uint4*)(vp + 8);
            acc[t] = wmma_bf16(Pf, vf, acc[t]);
        }
        __syncthreads();
    }

    // ---- finalize: divide by softmax sum, emit y bf16 [B][N][CO] ----
    if (lane < 16) invb[wid][lane] = 1.0f / s_run;
    __syncthreads();
    float inv[8];
#pragma unroll
    for (int r = 0; r < 8; ++r) inv[r] = invb[wid][r + 8 * hi];
#pragma unroll
    for (int t = 0; t < 8; ++t)
#pragma unroll
        for (int r = 0; r < 8; ++r)
            y[((size_t)(b * NN) + q0 + r + 8 * hi) * CO + t * 16 + col] =
                f2bf(acc[t][r] * inv[r]);
}

// ---------------- Kernel 5: output projection + bias + residual ---------------
__global__ void __launch_bounds__(128) k_oproj(
        const unsigned short* __restrict__ y, const unsigned short* __restrict__ wo,
        const float* __restrict__ wb, const float* __restrict__ x,
        float* __restrict__ out) {
    const int b    = blockIdx.y;
    const int n0   = blockIdx.x * 16;
    const int lane = threadIdx.x & 31;
    const int wid  = threadIdx.x >> 5;
    const int col  = lane & 15;
    const int hi   = lane >> 4;

    v8f zero = {0.f,0.f,0.f,0.f,0.f,0.f,0.f,0.f};
    v8f acc[4];
#pragma unroll
    for (int t = 0; t < 4; ++t) acc[t] = zero;

    const unsigned short* yrow = y + ((size_t)(b * NN) + n0 + col) * CO;
#pragma unroll
    for (int cc = 0; cc < 4; ++cc) {                    // K = 128 in 4 chunks of 32
        FragB by;
        const unsigned short* p = yrow + cc * 32 + hi * 16;
        by.q[0] = *(const uint4*)p; by.q[1] = *(const uint4*)(p + 8);
#pragma unroll
        for (int t = 0; t < 4; ++t) {
            const int c0 = wid * 64 + t * 16;           // Cin tile base (256 / (4 waves*4 tiles))
            const unsigned short* ap = wo + (size_t)(c0 + col) * CO + cc * 32 + hi * 16;
            FragB a;
            a.q[0] = *(const uint4*)ap; a.q[1] = *(const uint4*)(ap + 8);
            acc[t] = wmma_bf16(a, by, acc[t]);
        }
    }
#pragma unroll
    for (int t = 0; t < 4; ++t) {
        const int ce = wid * 64 + t * 16 + hi * 8;
        const float4 b0 = *(const float4*)(wb + ce);
        const float4 b1 = *(const float4*)(wb + ce + 4);
        float bias[8] = {b0.x,b0.y,b0.z,b0.w,b1.x,b1.y,b1.z,b1.w};
#pragma unroll
        for (int r = 0; r < 8; ++r) {
            size_t idx = ((size_t)(b * CIN) + ce + r) * NN + n0 + col;
            out[idx] = acc[t][r] + bias[r] + x[idx];
        }
    }
}

// ---------------- Host-side launch --------------------------------------------
extern "C" void kernel_launch(void* const* d_in, const int* in_sizes, int n_in,
                              void* d_out, int out_size, void* d_ws, size_t ws_size,
                              hipStream_t stream) {
    const float* x   = (const float*)d_in[0];
    const float* g_w = (const float*)d_in[1];
    const float* g_b = (const float*)d_in[2];
    const float* t_w = (const float*)d_in[3];
    const float* t_b = (const float*)d_in[4];
    const float* p_w = (const float*)d_in[5];
    const float* p_b = (const float*)d_in[6];
    const float* W_w = (const float*)d_in[7];
    const float* W_b = (const float*)d_in[8];
    float* out = (float*)d_out;

    char* ws = (char*)d_ws;
    size_t o = 0;
    unsigned short* xt   = (unsigned short*)(ws + o); o += (size_t)BB * NN * CIN * 2; // 8 MB
    unsigned short* qbuf = (unsigned short*)(ws + o); o += (size_t)BB * NN * CO  * 2; // 4 MB
    unsigned short* kbuf = (unsigned short*)(ws + o); o += (size_t)BB * NN * CO  * 2; // 4 MB
    unsigned short* vtbf = (unsigned short*)(ws + o); o += (size_t)BB * CO * NN  * 2; // 4 MB
    unsigned short* ybuf = (unsigned short*)(ws + o); o += (size_t)BB * NN * CO  * 2; // 4 MB
    unsigned short* wq   = (unsigned short*)(ws + o); o += (size_t)CO * CIN * 2;
    unsigned short* wk   = (unsigned short*)(ws + o); o += (size_t)CO * CIN * 2;
    unsigned short* wv   = (unsigned short*)(ws + o); o += (size_t)CO * CIN * 2;
    unsigned short* wo   = (unsigned short*)(ws + o); o += (size_t)CIN * CO * 2;

    k_xt   <<<dim3(NN / 32, CIN / 32, BB), dim3(32, 8), 0, stream>>>(x, xt);
    k_wcvt <<<dim3(128), dim3(256), 0, stream>>>(t_w, p_w, g_w, W_w, wq, wk, wv, wo);
    k_proj <<<dim3(NN / 16, BB), dim3(128), 0, stream>>>(xt, wq, wk, wv,
                                                         t_b, p_b, g_b,
                                                         qbuf, kbuf, vtbf);
    k_attn <<<dim3(BB * (NN / 16) / 4), dim3(128), 0, stream>>>(qbuf, kbuf, vtbf, ybuf);
    k_oproj<<<dim3(NN / 16, BB), dim3(128), 0, stream>>>(ybuf, wo, W_b, x, out);
}
